// RecommenderGCNModelAE_53678501266189
// MI455X (gfx1250) — matrix operand order repaired and checked by reading, not compiled, not run on hardware
//
#include <hip/hip_runtime.h>

// ---------------------------------------------------------------------------
// RecommenderGCNModelAE on MI455X (gfx1250), wave32 + WMMA bf16 + TDM.
//
// Memory-bound (~1.8 GB streamed vs ~47 GFLOP -> ~77us HBM floor @23.3TB/s):
//  * big fp32 matrices (X/ADJ, 256 MB) streamed via coalesced global_load_b128
//    + global_prefetch_b8,
//  * skinny [8192,64] B operand staged per 128-K tile into LDS by the Tensor
//    Data Mover (tensor_load_to_lds, double-buffered, s_wait_tensorcnt), with
//    descriptor padding (stride 65 dwords) to kill LDS bank conflicts,
//  * math in v_wmma_f32_16x16x32_bf16 (fp32 accumulate),
//  * split-K x8 with global_atomic_add_f32 for occupancy (4096 waves/GEMM).
// ---------------------------------------------------------------------------

typedef __attribute__((ext_vector_type(16))) __bf16 v16bf;
typedef __attribute__((ext_vector_type(8)))  float  v8f;
typedef unsigned int v4u __attribute__((ext_vector_type(4)));
typedef int          v8i __attribute__((ext_vector_type(8)));
typedef int          v4i __attribute__((ext_vector_type(4)));

#define NN      8192     // NU == NV == 8192
#define H1DIM   64
#define H2DIM   32
#define KSPLIT  8
#define KSTEP   32       // K per WMMA
#define KTILE   128      // K rows staged per TDM tile
#define LDSB_STRIDE 65   // 64 dwords/row + 1 pad dword (TDM pad_enable)

#if defined(__has_builtin)
#if __has_builtin(__builtin_amdgcn_tensor_load_to_lds)
#define HAS_TDM 1
#endif
#endif
#ifndef HAS_TDM
#define HAS_TDM 0
#endif

__device__ __forceinline__ v8f wmma_bf16(v16bf a, v16bf b, v8f c) {
  // (neg_a, A, neg_b, B, c_mod, C, reuse_a, reuse_b)
  return __builtin_amdgcn_wmma_f32_16x16x32_bf16(false, a, false, b,
                                                 (short)0, c, false, false);
}

// A-fragment: 16x32 tile of fp32 row-major A -> bf16 WMMA A layout.
// Contiguous k (not ISA table order; B matches): lane L holds row M=L%16; lanes 0-15 carry K={0..15},
// lanes 16-31 carry K={16..31}: one contiguous 64B segment per lane.
__device__ __forceinline__ v16bf load_a_frag(const float* __restrict__ A,
                                             long lda, int row0, long k0,
                                             int lane) {
  const int  m  = lane & 15;
  const long kh = (long)((lane >> 4) << 4);          // 0 or 16
  const float* p = A + (long)(row0 + m) * lda + k0 + kh;
  v16bf f;
#pragma unroll
  for (int e = 0; e < 8; ++e) f[e]     = (__bf16)p[e];
#pragma unroll
  for (int e = 0; e < 8; ++e) f[8 + e] = (__bf16)p[8 + e];
  return f;
}

// B-fragment from fp32 LDS tile (row stride LDSB_STRIDE dwords):
// element e -> K = kf + e, N = nl. Padded stride => 32 distinct banks.
__device__ __forceinline__ v16bf b_frag_lds(const float* __restrict__ bt,
                                            int nl, int kf) {
  v16bf b;
#pragma unroll
  for (int e = 0; e < 16; ++e)
    b[e] = (__bf16)bt[(kf + e) * LDSB_STRIDE + nl];
  return b;
}

#if HAS_TDM
// Issue one TDM 2-D tile load: B[k0..k0+KTILE)[0..64) fp32 -> LDS (padded).
// D# packing per CDNA5 ISA.
__device__ __forceinline__ void tdm_load_b_tile(const float* __restrict__ gsrc,
                                                float* ldst) {
  const unsigned long long ga = (unsigned long long)(const void*)gsrc;
  const unsigned int la = (unsigned int)(unsigned long long)(void*)ldst; // LDS offset = flat[31:0]
  v4u g0;
  g0[0] = 1u;                                            // count=1, user desc
  g0[1] = la;                                            // lds_addr
  g0[2] = (unsigned int)ga;                              // global_addr[31:0]
  g0[3] = ((unsigned int)(ga >> 32) & 0x01FFFFFFu)       // global_addr[56:32]
          | (2u << 30);                                  // type=2 (image)
  v8i g1;
  g1[0] = (int)((2u << 16)        // data_size = 4 bytes
              | (1u << 20)        // pad_enable
              | (5u << 22)        // pad_interval: every 64 dwords
              | (0u << 25));      // pad_amount: 1 dword
  g1[1] = (int)(64u << 16);       // [63:48]  tensor_dim0 lo16 = 64
  g1[2] = (int)(8192u << 16);     // [95:80]  tensor_dim1 lo16 = 8192
  g1[3] = (int)(64u << 16);       // [127:112] tile_dim0 = 64
  g1[4] = (int)(KTILE);           // [143:128] tile_dim1 = 128, tile_dim2 = 0
  g1[5] = 64;                     // [191:160] tensor_dim0_stride lo32 = 64
  g1[6] = 0;                      // dim0_stride hi | dim1_stride lo16
  g1[7] = 0;                      // tensor_dim1_stride (unused, 2-D tile)
  const v4i z4 = {0, 0, 0, 0};
#if __clang_major__ >= 23
  const v8i z8 = {0, 0, 0, 0, 0, 0, 0, 0};
  __builtin_amdgcn_tensor_load_to_lds(g0, g1, z4, z4, z8, 0);
#else
  __builtin_amdgcn_tensor_load_to_lds(g0, g1, z4, z4, 0);
#endif
}
#endif  // HAS_TDM

// ---------------------------------------------------------------------------
// C[M,64] += A[M,K] @ B[K,64]  (split-K, atomic fp32 accumulate)
// Block: 256 thr (8 waves); wave -> 16 rows x 64 cols.
// grid.x = M/128 row blocks, grid.y = KSPLIT.
// ---------------------------------------------------------------------------
__global__ void __launch_bounds__(256)
gemm_n64_splitk(const float* __restrict__ A, const float* __restrict__ B,
                float* __restrict__ C, long lda, int kPerSplit) {
  __shared__ __align__(16) float ldsB[2][KTILE * LDSB_STRIDE];  // 2 x 33280 B

  const int  tid  = threadIdx.x;
  const int  wave = tid >> 5;
  const int  lane = tid & 31;
  const int  row0 = blockIdx.x * 128 + wave * 16;
  const long kbase = (long)blockIdx.y * kPerSplit;
  const int  ntiles = kPerSplit / KTILE;

  const int nl   = lane & 15;
  const int koff = (lane >> 4) << 4;          // B frag: lanes 16-31 -> K+16
  const int mb   = (lane >> 4) << 3;          // C frag: lanes 16-31 -> M+8

  v8f acc0 = {}, acc1 = {}, acc2 = {}, acc3 = {};

#if HAS_TDM
  if (wave == 0) tdm_load_b_tile(B + kbase * H1DIM, &ldsB[0][0]);
#endif

  for (int t = 0; t < ntiles; ++t) {
    const float* bt = &ldsB[t & 1][0];

#if HAS_TDM
    if (wave == 0) __builtin_amdgcn_s_wait_tensorcnt((short)0);  // tile t ready
    __syncthreads();   // publish LDS tile t; also closes reads of buf (t+1)&1
    if (wave == 0 && t + 1 < ntiles)
      tdm_load_b_tile(B + (kbase + (long)(t + 1) * KTILE) * H1DIM,
                      &ldsB[(t + 1) & 1][0]);  // DMA overlaps compute below
#else
    __syncthreads();
    {  // cooperative fallback staging into the same padded layout
      const int r  = tid >> 1;           // 0..127
      const int c0 = (tid & 1) * 32;
      const float* src = B + (kbase + (long)t * KTILE + r) * H1DIM + c0;
      float* dst = (float*)bt + r * LDSB_STRIDE + c0;
#pragma unroll
      for (int j = 0; j < 32; ++j) dst[j] = src[j];
    }
    __syncthreads();
#endif

#pragma unroll
    for (int ks = 0; ks < KTILE; ks += KSTEP) {
      const long k0 = kbase + (long)t * KTILE + ks;
      v16bf afrag = load_a_frag(A, lda, row0, k0, lane);

      if (t + 1 < ntiles) {  // prefetch A one tile ahead (uniform guard)
        const float* pf =
            A + (long)(row0 + nl) * lda + k0 + ((lane >> 4) << 3) + KTILE;
        __builtin_prefetch(pf, 0, 0);            // global_prefetch_b8
      }

      const int kf = ks + koff;
      v16bf b0 = b_frag_lds(bt, nl +  0, kf);
      v16bf b1 = b_frag_lds(bt, nl + 16, kf);
      v16bf b2 = b_frag_lds(bt, nl + 32, kf);
      v16bf b3 = b_frag_lds(bt, nl + 48, kf);

      acc0 = wmma_bf16(afrag, b0, acc0);
      acc1 = wmma_bf16(afrag, b1, acc1);
      acc2 = wmma_bf16(afrag, b2, acc2);
      acc3 = wmma_bf16(afrag, b3, acc3);
    }
  }

  // Split-K reduction: fp32 global atomics into zeroed C.
#pragma unroll
  for (int r = 0; r < 8; ++r) {
    float* c = C + (long)(row0 + mb + r) * H1DIM + nl;
    __hip_atomic_fetch_add(c +  0, acc0[r], __ATOMIC_RELAXED, __HIP_MEMORY_SCOPE_AGENT);
    __hip_atomic_fetch_add(c + 16, acc1[r], __ATOMIC_RELAXED, __HIP_MEMORY_SCOPE_AGENT);
    __hip_atomic_fetch_add(c + 32, acc2[r], __ATOMIC_RELAXED, __HIP_MEMORY_SCOPE_AGENT);
    __hip_atomic_fetch_add(c + 48, acc3[r], __ATOMIC_RELAXED, __HIP_MEMORY_SCOPE_AGENT);
  }
}

// ---------------------------------------------------------------------------
// Z[M,32] = G[M,64] @ W2[64,32]   (tiny epilogue GEMM, K=64)
// ---------------------------------------------------------------------------
__global__ void __launch_bounds__(256)
zgemm_w2(const float* __restrict__ G, const float* __restrict__ W2,
         float* __restrict__ Z) {
  __shared__ __align__(64) __bf16 ldsB[H2DIM][H1DIM];   // [n][k]

  const int tid  = threadIdx.x;
  const int wave = tid >> 5;
  const int lane = tid & 31;
  const int row0 = blockIdx.x * 128 + wave * 16;

  {
    const int r  = tid >> 2;          // 0..63 : k
    const int c0 = (tid & 3) * 8;     // 0..24 : n start
#pragma unroll
    for (int j = 0; j < 8; ++j) ldsB[c0 + j][r] = (__bf16)W2[r * H2DIM + c0 + j];
  }
  __syncthreads();

  const int nl   = lane & 15;
  const int koff = (lane >> 4) << 4;
  v8f acc0 = {}, acc1 = {};
#pragma unroll
  for (int k0 = 0; k0 < H1DIM; k0 += KSTEP) {
    v16bf a  = load_a_frag(G, H1DIM, row0, k0, lane);
    v16bf b0 = *(const v16bf*)&ldsB[ 0 + nl][k0 + koff];
    v16bf b1 = *(const v16bf*)&ldsB[16 + nl][k0 + koff];
    acc0 = wmma_bf16(a, b0, acc0);
    acc1 = wmma_bf16(a, b1, acc1);
  }

  const int mb = (lane >> 4) << 3;
#pragma unroll
  for (int r = 0; r < 8; ++r) {
    float* z = Z + (long)(row0 + mb + r) * H2DIM + nl;
    z[0]  = acc0[r];
    z[16] = acc1[r];
  }
}

// ---------------------------------------------------------------------------
// OUT[8192,8192] = Zu @ Zv^T   (K=32, store-bound: 256 MB of output)
// Each wave: 16 rows x 64 cols. grid = (128 v-blocks, 64 u-blocks).
// ---------------------------------------------------------------------------
__global__ void __launch_bounds__(256)
recon_outer(const float* __restrict__ Zu, const float* __restrict__ Zv,
            float* __restrict__ Out) {
  const int tid  = threadIdx.x;
  const int wave = tid >> 5;
  const int lane = tid & 31;
  const int u0 = blockIdx.y * 128 + wave * 16;
  const int v0 = blockIdx.x * 64;

  v16bf a = load_a_frag(Zu, H2DIM, u0, 0, lane);

  const int nl   = lane & 15;
  const int koff = (lane >> 4) << 4;
  v8f acc[4];
#pragma unroll
  for (int nt = 0; nt < 4; ++nt) {
    // B[k][n] = Zv[v0 + nt*16 + n][k]: 16 contiguous floats (64B) per lane.
    const float* pv = Zv + (long)(v0 + nt * 16 + nl) * H2DIM + koff;
    v16bf b;
#pragma unroll
    for (int e = 0; e < 16; ++e) b[e] = (__bf16)pv[e];
    v8f z = {};
    acc[nt] = wmma_bf16(a, b, z);
  }

  const int mb = (lane >> 4) << 3;
#pragma unroll
  for (int r = 0; r < 8; ++r) {
    float* o = Out + (long)(u0 + mb + r) * (long)NN + v0 + nl;
    o[0]  = acc[0][r];
    o[16] = acc[1][r];
    o[32] = acc[2][r];
    o[48] = acc[3][r];
  }
}

// --------------------------- small utility kernels -------------------------
__global__ void zero_f32(float* __restrict__ p, long n) {
  long i = (long)blockIdx.x * blockDim.x + threadIdx.x;
  const long stride = (long)gridDim.x * blockDim.x;
  for (; i < n; i += stride) p[i] = 0.0f;
}

__global__ void relu_f32(float* __restrict__ p, long n) {
  long i = (long)blockIdx.x * blockDim.x + threadIdx.x;
  const long stride = (long)gridDim.x * blockDim.x;
  for (; i < n; i += stride) { float v = p[i]; p[i] = v > 0.0f ? v : 0.0f; }
}

// ---------------------------------------------------------------------------
extern "C" void kernel_launch(void* const* d_in, const int* in_sizes, int n_in,
                              void* d_out, int out_size, void* d_ws, size_t ws_size,
                              hipStream_t stream) {
  (void)in_sizes; (void)n_in; (void)out_size; (void)ws_size;

  const float* x_user   = (const float*)d_in[0];
  const float* adj_user = (const float*)d_in[1];
  const float* w1_user  = (const float*)d_in[2];
  const float* w2_user  = (const float*)d_in[3];
  const float* x_item   = (const float*)d_in[4];
  const float* adj_item = (const float*)d_in[5];
  const float* w1_item  = (const float*)d_in[6];
  const float* w2_item  = (const float*)d_in[7];
  float* out = (float*)d_out;

  const long NH1 = (long)NN * H1DIM;   // 524288
  const long NH2 = (long)NN * H2DIM;   // 262144

  float* ws  = (float*)d_ws;
  float* H0u = ws;                // [NN,64]
  float* H1u = H0u + NH1;         // [NN,64]
  float* Gu  = H1u + NH1;         // [NN,64]
  float* Zu  = Gu  + NH1;         // [NN,32]
  float* H0i = Zu  + NH2;
  float* H1i = H0i + NH1;
  float* Gi  = H1i + NH1;
  float* Zi  = Gi  + NH1;
  const long totalScratch = 2 * (3 * NH1 + NH2);   // ~14 MB of fp32

  // Split-K atomic accumulation needs zeroed accumulators every call.
  zero_f32<<<2048, 256, 0, stream>>>(ws, totalScratch);

  const dim3 gGemm(NN / 128, KSPLIT);
  const int  kPerSplit = NN / KSPLIT;   // 1024

  // ---- user branch ----
  gemm_n64_splitk<<<gGemm, 256, 0, stream>>>(x_user,   w1_user, H0u, NN, kPerSplit);
  gemm_n64_splitk<<<gGemm, 256, 0, stream>>>(adj_user, H0u,     H1u, NN, kPerSplit);
  relu_f32<<<512, 256, 0, stream>>>(H1u, NH1);
  gemm_n64_splitk<<<gGemm, 256, 0, stream>>>(adj_user, H1u,     Gu,  NN, kPerSplit);
  zgemm_w2<<<NN / 128, 256, 0, stream>>>(Gu, w2_user, Zu);

  // ---- item branch ----
  gemm_n64_splitk<<<gGemm, 256, 0, stream>>>(x_item,   w1_item, H0i, NN, kPerSplit);
  gemm_n64_splitk<<<gGemm, 256, 0, stream>>>(adj_item, H0i,     H1i, NN, kPerSplit);
  relu_f32<<<512, 256, 0, stream>>>(H1i, NH1);
  gemm_n64_splitk<<<gGemm, 256, 0, stream>>>(adj_item, H1i,     Gi,  NN, kPerSplit);
  zgemm_w2<<<NN / 128, 256, 0, stream>>>(Gi, w2_item, Zi);

  // ---- decoder: OUT = Zu @ Zi^T ----
  recon_outer<<<dim3(NN / 64, NN / 128), 256, 0, stream>>>(Zu, Zi, out);
}
